// SparseExpertRouter_84963043049701
// MI455X (gfx1250) — matrix operand order repaired and checked by reading, not compiled, not run on hardware
//
#include <hip/hip_runtime.h>
#include <hip/hip_bf16.h>

// ---------------- problem dims (fixed by reference) ----------------
#define T_TOK   8192   // B*S = 4*2048
#define DIM     1024
#define HID     4096
#define RDIM    512
#define NEXP    8

typedef unsigned short ushort_t;
typedef __attribute__((ext_vector_type(16))) __bf16 v16bf;
typedef __attribute__((ext_vector_type(8)))  float  v8f;

// ---------------- scalar helpers ----------------
__device__ __forceinline__ ushort_t f2bf(float f) {
    unsigned u = __builtin_bit_cast(unsigned, f);
    unsigned r = u + 0x7FFFu + ((u >> 16) & 1u);   // RNE
    return (ushort_t)(r >> 16);
}
__device__ __forceinline__ float bf2f(ushort_t h) {
    unsigned u = ((unsigned)h) << 16;
    return __builtin_bit_cast(float, u);
}
__device__ __forceinline__ float gelu_exact(float x) {
    return 0.5f * x * (1.0f + erff(x * 0.70710678118654752f));
}

// Load a 32-byte bf16 fragment (two 16B chunks) and bit-cast to v16bf.
__device__ __forceinline__ v16bf load_frag(const ushort_t* p0, const ushort_t* p1) {
    struct Q { uint4 a, b; } q;
    q.a = *reinterpret_cast<const uint4*>(p0);
    q.b = *reinterpret_cast<const uint4*>(p1);
    return __builtin_bit_cast(v16bf, q);
}

// ---------------- register-blocked, software-pipelined WMMA core ----------------
// 2 m-frags x 4 n-frags per wave; K unrolled by 64 with ping-pong fragment buffers
// so loads for stage k+32 are in flight while WMMAs consume stage k (partial
// s_wait_loadcnt instead of a full drain per step). Requires K % 64 == 0.
__device__ __forceinline__ void gemm_core(const ushort_t* const* Ar,   // [2] row pointers
                                          const ushort_t* const* Wr,   // [4] B-col pointers
                                          int K, int aOff, int bOff,
                                          v8f acc[2][4]) {
    v16bf a0[2], b0[4], a1[2], b1[4];
    // prologue: stage-0 fragments at k=0
    #pragma unroll
    for (int i = 0; i < 2; ++i)
        a0[i] = load_frag(Ar[i] + aOff, Ar[i] + aOff + 8);
    #pragma unroll
    for (int j = 0; j < 4; ++j)
        b0[j] = load_frag(Wr[j] + bOff, Wr[j] + bOff + 8);

    for (int k0 = 0; k0 < K; k0 += 64) {
        // issue stage-1 loads (k0+32) before computing on stage 0
        #pragma unroll
        for (int i = 0; i < 2; ++i)
            a1[i] = load_frag(Ar[i] + k0 + 32 + aOff, Ar[i] + k0 + 32 + aOff + 8);
        #pragma unroll
        for (int j = 0; j < 4; ++j) {
            __builtin_prefetch(Wr[j] + k0 + 512, 0, 1);  // global_prefetch_b8 on weight stream
            b1[j] = load_frag(Wr[j] + k0 + 32 + bOff, Wr[j] + k0 + 32 + bOff + 8);
        }
        // compute stage 0
        #pragma unroll
        for (int i = 0; i < 2; ++i)
            #pragma unroll
            for (int j = 0; j < 4; ++j)
                acc[i][j] = __builtin_amdgcn_wmma_f32_16x16x32_bf16(
                    false, a0[i], false, b0[j], (short)0, acc[i][j], false, false);
        // refill stage 0 with k0+64 (uniform guard; EXEC untouched)
        if (k0 + 64 < K) {
            #pragma unroll
            for (int i = 0; i < 2; ++i)
                a0[i] = load_frag(Ar[i] + k0 + 64 + aOff, Ar[i] + k0 + 64 + aOff + 8);
            #pragma unroll
            for (int j = 0; j < 4; ++j)
                b0[j] = load_frag(Wr[j] + k0 + 64 + bOff, Wr[j] + k0 + 64 + bOff + 8);
        }
        // compute stage 1
        #pragma unroll
        for (int i = 0; i < 2; ++i)
            #pragma unroll
            for (int j = 0; j < 4; ++j)
                acc[i][j] = __builtin_amdgcn_wmma_f32_16x16x32_bf16(
                    false, a1[i], false, b1[j], (short)0, acc[i][j], false, false);
    }
}

// ---------------- utility kernels ----------------
__global__ void k_zero(float* __restrict__ out, long n, int* __restrict__ cnt, int nc) {
    long i = (long)blockIdx.x * blockDim.x + threadIdx.x;
    long stride = (long)gridDim.x * blockDim.x;
    for (long j = i; j < n; j += stride) out[j] = 0.0f;
    if (i < nc) cnt[i] = 0;
}

__global__ void k_cast_bf16(const float* __restrict__ src, ushort_t* __restrict__ dst, long n) {
    long i = (long)blockIdx.x * blockDim.x + threadIdx.x;
    long stride = (long)gridDim.x * blockDim.x;
    for (long j = i; j < n; j += stride) dst[j] = f2bf(src[j]);
}

// ---------------- WMMA GEMM + bias + GELU -> bf16 ----------------
// C[m,n] = GELU( sum_k A[row(m),k] * W[n,k] + bias[n] )
// grid: (ceil(cnt/64), N/256, E-or-1), block 256 = 8 wave32s in a 2(m) x 4(n) grid;
// each wave computes 32x64 via a 2x4 register tile of 16x16 WMMA accumulators.
__global__ void __launch_bounds__(256)
k_gemm_gelu(const ushort_t* __restrict__ A, int lda,
            const ushort_t* __restrict__ Wg, long wStride,
            const float* __restrict__ biasG, int biasStride,
            const int* __restrict__ gatherG, int gatherStride,
            const int* __restrict__ cntArr, int defCnt,
            const int* __restrict__ baseArr,
            ushort_t* __restrict__ Y, int ldy,
            int K) {
    const int e = blockIdx.z;
    const ushort_t* W   = Wg + (long)e * wStride;
    const float* bias   = biasG + (long)e * biasStride;
    const int* gather   = gatherG ? (gatherG + (long)e * gatherStride) : nullptr;
    const int  cnt      = cntArr ? cntArr[e] : defCnt;
    const int  m0       = blockIdx.x * 64;
    if (m0 >= cnt) return;                       // uniform exit -> EXEC stays all-1 for WMMA
    const int rowBase   = baseArr ? baseArr[e] : 0;
    const int n0        = blockIdx.y * 256;

    const int wave = threadIdx.x >> 5;           // 0..7
    const int lane = threadIdx.x & 31;
    const int mi = m0 + (wave & 1) * 32;         // 2 m-waves x 32 rows
    const int ni = n0 + (wave >> 1) * 64;        // 4 n-waves x 64 cols
    const int l15 = lane & 15;
    const bool lo = (lane < 16);
    const int aOff = lo ? 0 : 16;                // lanes 0-15: K 0..15 ; 16-31: K 16..31
    const int bOff = lo ? 0 : 16;                // lanes 0-15: K 0..15 ; 16-31: K 16..31

    const ushort_t* Ar[2];
    #pragma unroll
    for (int i = 0; i < 2; ++i) {
        int arow = mi + i * 16 + l15; if (arow >= cnt) arow = cnt - 1;   // clamp keeps EXEC full
        const int asrc = gather ? gather[arow] : arow;
        Ar[i] = A + (long)asrc * lda;
    }
    const ushort_t* Wr[4];
    #pragma unroll
    for (int j = 0; j < 4; ++j)
        Wr[j] = W + (long)(ni + j * 16 + l15) * K;

    v8f acc[2][4] = {};
    gemm_core(Ar, Wr, K, aOff, bOff, acc);

    // epilogue: lane holds col = l15 (+16j), rows rbase..rbase+7
    #pragma unroll
    for (int i = 0; i < 2; ++i) {
        const int rbase = mi + i * 16 + (lo ? 0 : 8);
        #pragma unroll
        for (int j = 0; j < 4; ++j) {
            const int c = ni + j * 16 + l15;
            const float bv = bias[c];
            #pragma unroll
            for (int q = 0; q < 8; ++q) {
                const int row = rbase + q;
                if (row < cnt) {
                    float v = gelu_exact(acc[i][j][q] + bv);
                    Y[(long)(rowBase + row) * ldy + c] = f2bf(v);
                }
            }
        }
    }
}

// ---------------- WMMA GEMM + bias, scaled atomic scatter ----------------
// out[tok, n] += w_tok * ( sum_k He[base+m,k]*W[n,k] + bias[n] )
__global__ void __launch_bounds__(256)
k_gemm_scatter(const ushort_t* __restrict__ He, int lda,
               const ushort_t* __restrict__ Wg, long wStride,
               const float* __restrict__ biasG, int biasStride,
               const int* __restrict__ tokG, const float* __restrict__ wtG,
               const int* __restrict__ cntArr,
               const int* __restrict__ baseArr,
               float* __restrict__ out, int ldo,
               int K) {
    const int e = blockIdx.z;
    const ushort_t* W = Wg + (long)e * wStride;
    const float* bias = biasG + (long)e * biasStride;
    const int* toks   = tokG + (long)e * T_TOK;
    const float* wts  = wtG + (long)e * T_TOK;
    const int cnt     = cntArr[e];
    const int m0      = blockIdx.x * 64;
    if (m0 >= cnt) return;
    const int rowBase = baseArr[e];
    const int n0      = blockIdx.y * 256;

    const int wave = threadIdx.x >> 5;
    const int lane = threadIdx.x & 31;
    const int mi = m0 + (wave & 1) * 32;
    const int ni = n0 + (wave >> 1) * 64;
    const int l15 = lane & 15;
    const bool lo = (lane < 16);
    const int aOff = lo ? 0 : 16;
    const int bOff = lo ? 0 : 16;

    const ushort_t* Ar[2];
    #pragma unroll
    for (int i = 0; i < 2; ++i) {
        int arow = mi + i * 16 + l15; if (arow >= cnt) arow = cnt - 1;
        Ar[i] = He + (long)(rowBase + arow) * lda;
    }
    const ushort_t* Wr[4];
    #pragma unroll
    for (int j = 0; j < 4; ++j)
        Wr[j] = W + (long)(ni + j * 16 + l15) * K;

    v8f acc[2][4] = {};
    gemm_core(Ar, Wr, K, aOff, bOff, acc);

    #pragma unroll
    for (int i = 0; i < 2; ++i) {
        const int rbase = mi + i * 16 + (lo ? 0 : 8);
        #pragma unroll
        for (int j = 0; j < 4; ++j) {
            const int c = ni + j * 16 + l15;
            const float bv = bias[c];
            #pragma unroll
            for (int q = 0; q < 8; ++q) {
                const int row = rbase + q;
                if (row < cnt) {
                    const int tok = toks[row];
                    const float w = wts[row];
                    atomicAdd(&out[(long)tok * ldo + c], w * (acc[i][j][q] + bv));
                }
            }
        }
    }
}

// ---------------- router stage 2: logits -> softmax -> top-2 -> dispatch ----------------
__global__ void k_router_top2(const ushort_t* __restrict__ h,   // [T, R] bf16
                              const float* __restrict__ Wr2,    // [E, R]
                              const float* __restrict__ br2,    // [E]
                              int* __restrict__ cnt,            // [E]
                              int* __restrict__ toke,           // [E, T]
                              float* __restrict__ wte) {        // [E, T]
    const int t = blockIdx.x * blockDim.x + threadIdx.x;
    if (t >= T_TOK) return;
    const ushort_t* hrow = h + (long)t * RDIM;

    float lg[NEXP];
    #pragma unroll
    for (int e = 0; e < NEXP; ++e) {
        const float* wr = Wr2 + e * RDIM;
        float s = br2[e];
        for (int r = 0; r < RDIM; r += 4) {
            s += bf2f(hrow[r + 0]) * wr[r + 0];
            s += bf2f(hrow[r + 1]) * wr[r + 1];
            s += bf2f(hrow[r + 2]) * wr[r + 2];
            s += bf2f(hrow[r + 3]) * wr[r + 3];
        }
        lg[e] = s;
    }
    // softmax
    float m = lg[0];
    #pragma unroll
    for (int e = 1; e < NEXP; ++e) m = fmaxf(m, lg[e]);
    float sum = 0.0f;
    float p[NEXP];
    #pragma unroll
    for (int e = 0; e < NEXP; ++e) { p[e] = __expf(lg[e] - m); sum += p[e]; }
    const float inv = 1.0f / sum;
    #pragma unroll
    for (int e = 0; e < NEXP; ++e) p[e] *= inv;
    // top-2 (distinct)
    int i1 = 0; float p1 = p[0];
    #pragma unroll
    for (int e = 1; e < NEXP; ++e) if (p[e] > p1) { p1 = p[e]; i1 = e; }
    int i2 = (i1 == 0) ? 1 : 0; float p2 = p[i2];
    #pragma unroll
    for (int e = 0; e < NEXP; ++e) if (e != i1 && p[e] > p2) { p2 = p[e]; i2 = e; }
    // dispatch
    int pos1 = atomicAdd(&cnt[i1], 1);
    toke[(long)i1 * T_TOK + pos1] = t;
    wte [(long)i1 * T_TOK + pos1] = p1;
    int pos2 = atomicAdd(&cnt[i2], 1);
    toke[(long)i2 * T_TOK + pos2] = t;
    wte [(long)i2 * T_TOK + pos2] = p2;
}

__global__ void k_prefix(const int* __restrict__ cnt, int* __restrict__ base) {
    if (threadIdx.x == 0 && blockIdx.x == 0) {
        int s = 0;
        for (int e = 0; e < NEXP; ++e) { base[e] = s; s += cnt[e]; }
    }
}

// ---------------- host-side orchestration ----------------
static inline size_t alignup(size_t x) { return (x + 255) & ~(size_t)255; }

extern "C" void kernel_launch(void* const* d_in, const int* in_sizes, int n_in,
                              void* d_out, int out_size, void* d_ws, size_t ws_size,
                              hipStream_t stream) {
    const float* x   = (const float*)d_in[0];   // [T, D]
    const float* Wr1 = (const float*)d_in[1];   // [R, D]
    const float* br1 = (const float*)d_in[2];   // [R]
    const float* Wr2 = (const float*)d_in[3];   // [E, R]
    const float* br2 = (const float*)d_in[4];   // [E]
    const float* W1  = (const float*)d_in[5];   // [E, H, D]
    const float* b1  = (const float*)d_in[6];   // [E, H]
    const float* W2  = (const float*)d_in[7];   // [E, D, H]
    const float* b2  = (const float*)d_in[8];   // [E, D]
    float* out = (float*)d_out;                 // [T, D]

    // workspace carve-up (bf16 weight/activation copies: whole bf16 weight set ~128MB -> L2-resident)
    char* ws = (char*)d_ws;
    size_t o = 0;
    ushort_t* xb   = (ushort_t*)(ws + o); o += alignup((size_t)T_TOK * DIM * 2);
    ushort_t* wr1b = (ushort_t*)(ws + o); o += alignup((size_t)RDIM * DIM * 2);
    ushort_t* w1b  = (ushort_t*)(ws + o); o += alignup((size_t)NEXP * HID * DIM * 2);
    ushort_t* w2b  = (ushort_t*)(ws + o); o += alignup((size_t)NEXP * DIM * HID * 2);
    ushort_t* hb   = (ushort_t*)(ws + o); o += alignup((size_t)T_TOK * RDIM * 2);
    ushort_t* heg  = (ushort_t*)(ws + o); o += alignup((size_t)2 * T_TOK * HID * 2);
    int*      toke = (int*)     (ws + o); o += alignup((size_t)NEXP * T_TOK * 4);
    float*    wte  = (float*)   (ws + o); o += alignup((size_t)NEXP * T_TOK * 4);
    int*      cnt  = (int*)     (ws + o); o += alignup((size_t)NEXP * 4);
    int*      base = (int*)     (ws + o); o += alignup((size_t)NEXP * 4);
    (void)ws_size; (void)n_in; (void)in_sizes; (void)out_size;

    // 1) zero output + counters
    k_zero<<<4096, 256, 0, stream>>>(out, (long)T_TOK * DIM, cnt, NEXP);

    // 2) fp32 -> bf16 casts
    k_cast_bf16<<<4096, 256, 0, stream>>>(x,   xb,   (long)T_TOK * DIM);
    k_cast_bf16<<<2048, 256, 0, stream>>>(Wr1, wr1b, (long)RDIM * DIM);
    k_cast_bf16<<<8192, 256, 0, stream>>>(W1,  w1b,  (long)NEXP * HID * DIM);
    k_cast_bf16<<<8192, 256, 0, stream>>>(W2,  w2b,  (long)NEXP * DIM * HID);

    // 3) router GEMM: h = GELU(x @ Wr1^T + br1)   [T, R]
    {
        dim3 grid(T_TOK / 64, RDIM / 256, 1);
        k_gemm_gelu<<<grid, 256, 0, stream>>>(xb, DIM,
                                              wr1b, 0, br1, 0,
                                              nullptr, 0, nullptr, T_TOK, nullptr,
                                              hb, RDIM, DIM);
    }

    // 4) router: logits -> softmax -> top-2 -> expert gather lists
    k_router_top2<<<(T_TOK + 255) / 256, 256, 0, stream>>>(hb, Wr2, br2, cnt, toke, wte);

    // 5) compact slot bases
    k_prefix<<<1, 32, 0, stream>>>(cnt, base);

    // 6) expert GEMM1: he = GELU(x_gathered @ W1[e]^T + b1[e]) -> heg (compact rows)
    {
        dim3 grid(T_TOK / 64, HID / 256, NEXP);
        k_gemm_gelu<<<grid, 256, 0, stream>>>(xb, DIM,
                                              w1b, (long)HID * DIM, b1, HID,
                                              toke, T_TOK, cnt, 0, base,
                                              heg, HID, DIM);
    }

    // 7) expert GEMM2: out[tok] += w * (he @ W2[e]^T + b2[e])
    {
        dim3 grid(T_TOK / 64, DIM / 256, NEXP);
        k_gemm_scatter<<<grid, 256, 0, stream>>>(heg, HID,
                                                 w2b, (long)DIM * HID, b2, DIM,
                                                 toke, wte, cnt, base,
                                                 out, DIM, HID);
    }
}
